// Pointnet2Encoder_15573551415826
// MI455X (gfx1250) — matrix operand order repaired and checked by reading, not compiled, not run on hardware
//
#include <hip/hip_runtime.h>

typedef __attribute__((ext_vector_type(16))) __bf16 v16bf;
typedef __attribute__((ext_vector_type(8)))  float  v8f;

union FragU {                  // 32B: two b128 loads -> one WMMA operand
    uint4  q[2];
    v16bf  v;
};

static __device__ __forceinline__ unsigned pack_bf16(float a, float b)
{
    const unsigned short lo = __builtin_bit_cast(unsigned short, (__bf16)a);
    const unsigned short hi = __builtin_bit_cast(unsigned short, (__bf16)b);
    return ((unsigned)hi << 16) | lo;
}

// ---------------------------------------------------------------------------
// Furthest-point sampling: one persistent 1024-thread workgroup (32 wave32s).
// dist[] (<=256KB) stays L2 resident. Per step: strided min-update + two-level
// argmax (in-wave shuffle tree, then one wave combines 32 partials via LDS).
// ---------------------------------------------------------------------------
__global__ __launch_bounds__(1024) void fps_kernel(
    const float* __restrict__ xyz, int N, int npoint,
    float* __restrict__ dist, int* __restrict__ fidx)
{
    __shared__ float swv[32];
    __shared__ int   swi[32];
    __shared__ int   sbcast;

    const int tid  = threadIdx.x;
    const int lane = tid & 31;
    const int wid  = tid >> 5;

    for (int n = tid; n < N; n += 1024) dist[n] = 1e10f;
    __syncthreads();

    int cur = 0;
    for (int i = 0; i < npoint; ++i) {
        cur = __builtin_amdgcn_readfirstlane(cur);
        if (tid == 0) fidx[i] = cur;
        const float fx = xyz[cur * 3 + 0];
        const float fy = xyz[cur * 3 + 1];
        const float fz = xyz[cur * 3 + 2];

        float bestv = -1.0f;
        int   besti = 0;
        for (int n = tid; n < N; n += 1024) {
            const float dx = xyz[n * 3 + 0] - fx;
            const float dy = xyz[n * 3 + 1] - fy;
            const float dz = xyz[n * 3 + 2] - fz;
            const float d  = dx * dx + dy * dy + dz * dz;
            const float dd = fminf(dist[n], d);
            dist[n] = dd;
            if (dd > bestv) { bestv = dd; besti = n; }
        }
        #pragma unroll
        for (int off = 16; off > 0; off >>= 1) {
            const float ov = __shfl_down(bestv, off, 32);
            const int   oi = __shfl_down(besti, off, 32);
            if (ov > bestv || (ov == bestv && oi < besti)) { bestv = ov; besti = oi; }
        }
        if (lane == 0) { swv[wid] = bestv; swi[wid] = besti; }
        __syncthreads();
        if (wid == 0) {
            bestv = swv[lane];
            besti = swi[lane];
            #pragma unroll
            for (int off = 16; off > 0; off >>= 1) {
                const float ov = __shfl_down(bestv, off, 32);
                const int   oi = __shfl_down(besti, off, 32);
                if (ov > bestv || (ov == bestv && oi < besti)) { bestv = ov; besti = oi; }
            }
            if (lane == 0) sbcast = besti;
        }
        __syncthreads();
        cur = sbcast;
    }
}

// ---------------------------------------------------------------------------
__global__ void gather_xyz_kernel(const float* __restrict__ xyz,
                                  const int* __restrict__ fidx,
                                  float* __restrict__ nxyz, int P)
{
    const int i = blockIdx.x * 256 + threadIdx.x;
    if (i < P) {
        const int j = fidx[i];
        nxyz[i * 3 + 0] = xyz[j * 3 + 0];
        nxyz[i * 3 + 1] = xyz[j * 3 + 1];
        nxyz[i * 3 + 2] = xyz[j * 3 + 2];
    }
}

// ---------------------------------------------------------------------------
// Ball query: first nsample hits in index order, pad with first hit (0 if none).
// Early-exit keeps the average scan short; prefetch hides the stream latency.
// ---------------------------------------------------------------------------
__global__ void ball_query_kernel(const float* __restrict__ cxyz,
                                  const float* __restrict__ xyz,
                                  int P, int N, float r2, int nsample,
                                  int* __restrict__ gidx)
{
    const int p = blockIdx.x * 128 + threadIdx.x;
    if (p >= P) return;
    const float cx = cxyz[p * 3 + 0];
    const float cy = cxyz[p * 3 + 1];
    const float cz = cxyz[p * 3 + 2];
    int* out = gidx + (size_t)p * nsample;
    int cnt = 0;
    for (int j = 0; j < N && cnt < nsample; ++j) {
        if ((j & 63) == 0 && j + 256 < N)
            __builtin_prefetch(&xyz[(j + 256) * 3], 0, 0);  // global_prefetch_b8
        const float dx = xyz[j * 3 + 0] - cx;
        const float dy = xyz[j * 3 + 1] - cy;
        const float dz = xyz[j * 3 + 2] - cz;
        if (dx * dx + dy * dy + dz * dz < r2) out[cnt++] = j;
    }
    const int pad = (cnt > 0) ? out[0] : 0;
    for (; cnt < nsample; ++cnt) out[cnt] = pad;
}

// ---------------------------------------------------------------------------
__global__ void zero_kernel(float* __restrict__ p, size_t n)
{
    for (size_t i = blockIdx.x * (size_t)blockDim.x + threadIdx.x; i < n;
         i += (size_t)gridDim.x * blockDim.x)
        p[i] = 0.0f;
}

// ---------------------------------------------------------------------------
// Pre-pack W into bf16 pairs laid out in exact WMMA A-fragment order:
//   t = ((mt*nkt + kt)*32 + lane)*8 + j2
// so the GEMM loads each A fragment as two contiguous b128 reads.
// ---------------------------------------------------------------------------
__global__ void pack_w_kernel(const float* __restrict__ W,
                              unsigned* __restrict__ wpack,
                              int total, int Cin, int nkt)
{
    for (int t = blockIdx.x * blockDim.x + threadIdx.x; t < total;
         t += gridDim.x * blockDim.x) {
        const int j2    = t & 7;
        const int lane  = (t >> 3) & 31;
        const int mtkt  = t >> 8;
        const int kt    = mtkt % nkt;
        const int mt    = mtkt / nkt;
        const int mrow  = mt * 16 + (lane & 15);
        const int kb8   = (lane >> 4) << 3;
        const int kbase = ((j2 < 4) ? (2 * j2) : (16 + 2 * (j2 - 4))) + kb8;
        const int kg    = kt * 32 + kbase;
        const float* wr = W + (size_t)mrow * Cin;
        const float v0 = (kg     < Cin) ? wr[kg]     : 0.f;
        const float v1 = (kg + 1 < Cin) ? wr[kg + 1] : 0.f;
        wpack[t] = pack_bf16(v0, v1);
    }
}

// ---------------------------------------------------------------------------
static __device__ __forceinline__ float g_elem(
    int k, int j, const float* __restrict__ xyz,
    float cx, float cy, float cz,
    const float* __restrict__ feats, int N, int Cin)
{
    if (k >= Cin) return 0.f;
    if (k == 0)   return xyz[j * 3 + 0] - cx;
    if (k == 1)   return xyz[j * 3 + 1] - cy;
    if (k == 2)   return xyz[j * 3 + 2] - cz;
    return feats[(size_t)(k - 3) * N + j];
}

// ---------------------------------------------------------------------------
// Fused group + GEMM (v_wmma_f32_16x16x32_bf16) + BN + ReLU + max-pool.
// One block (4 waves) per 16-column tile of the flattened (P*S) dimension;
// S in {32,64} is a multiple of 16 so each tile maps to a single group p.
// LDS g-tile is stored as packed bf16 pairs in B-fragment order:
//   sBp[col*16 + pair]  ->  each lane's fragment = 8 contiguous dwords
//   -> two ds_load_b128 per B fragment, two global_load_b128 per A fragment.
// ---------------------------------------------------------------------------
template <int MT>
__global__ __launch_bounds__(128) void sa_gemm_pool_kernel(
    const unsigned* __restrict__ wpack,   // A fragments, pre-packed bf16 pairs
    const float* __restrict__ gamma,      // [O]
    const float* __restrict__ beta,       // [O]
    const float* __restrict__ xyz,        // [N, 3]
    const float* __restrict__ cxyz,       // [P, 3]
    const float* __restrict__ feats,      // [Cin-3, N]
    const int*   __restrict__ gidx,       // [P*S]
    float*       __restrict__ outf,       // [O, P] zeroed; atomic-max on bits
    int Cin, int nkt, int P, int S, int N)
{
    __shared__ __align__(16) unsigned sBp[256];   // 16 cols x 16 bf16-pairs
    __shared__ int sJ[16];

    const int tid  = threadIdx.x;
    const int lane = tid & 31;
    const int wave = tid >> 5;
    const int col0 = blockIdx.x * 16;
    const int p    = col0 / S;                    // constant across the tile

    if (tid < 16) sJ[tid] = gidx[col0 + tid];
    __syncthreads();

    const float cx = cxyz[p * 3 + 0];
    const float cy = cxyz[p * 3 + 1];
    const float cz = cxyz[p * 3 + 2];

    const v8f vzero = {0.f, 0.f, 0.f, 0.f, 0.f, 0.f, 0.f, 0.f};
    v8f acc[MT];
    #pragma unroll
    for (int i = 0; i < MT; ++i) acc[i] = vzero;

    for (int kt = 0; kt < nkt; ++kt) {
        // Stage the 32x16 g-tile: 256 packed pairs, 2 per thread
        // (adjacent addresses -> merged ds_store_b64).
        #pragma unroll
        for (int it = 0; it < 2; ++it) {
            const int u   = tid * 2 + it;         // = col*16 + pair
            const int col = u >> 4;
            const int k   = kt * 32 + ((u & 3) << 1) + ((u & 4) << 2) + (u & 8);
            const int j   = sJ[col];
            const float v0 = g_elem(k,     j, xyz, cx, cy, cz, feats, N, Cin);
            const float v1 = g_elem(k + 1, j, xyz, cx, cy, cz, feats, N, Cin);
            sBp[u] = pack_bf16(v0, v1);
        }
        __syncthreads();

        // B fragment: 8 contiguous dwords per lane -> 2x ds_load_b128.
        FragU fb;
        {
            const uint4* bp = (const uint4*)(sBp + ((lane & 15) << 4) + ((lane >> 4) << 3));
            fb.q[0] = bp[0];
            fb.q[1] = bp[1];
        }

        #pragma unroll
        for (int mi = 0; mi < MT; ++mi) {
            const int mt = mi * 4 + wave;
            // A fragment: 8 contiguous dwords per lane -> 2x global_load_b128.
            FragU fa;
            const uint4* ap =
                (const uint4*)(wpack + (((size_t)(mt * nkt + kt) * 32 + lane) << 3));
            fa.q[0] = ap[0];
            fa.q[1] = ap[1];
            acc[mi] = __builtin_amdgcn_wmma_f32_16x16x32_bf16(
                false, fa.v, false, fb.v, (short)0, acc[mi], false, false);
        }
        __syncthreads();
    }

    // Epilogue: BN + ReLU, max over the tile's 16 columns (all same p),
    // then one atomic per row. Post-ReLU >= 0 so uint-bit max is exact.
    #pragma unroll
    for (int mi = 0; mi < MT; ++mi) {
        const int mbase = (mi * 4 + wave) * 16 + ((lane >> 4) << 3);
        #pragma unroll
        for (int r = 0; r < 8; ++r) {
            const int m = mbase + r;
            float v = fmaxf(acc[mi][r] * gamma[m] + beta[m], 0.f);
            #pragma unroll
            for (int off = 1; off < 16; off <<= 1)
                v = fmaxf(v, __shfl_xor(v, off, 16));
            if ((lane & 15) == 0)
                atomicMax((unsigned int*)&outf[(size_t)m * P + p], __float_as_uint(v));
        }
    }
}

// ---------------------------------------------------------------------------
extern "C" void kernel_launch(void* const* d_in, const int* in_sizes, int n_in,
                              void* d_out, int out_size, void* d_ws, size_t ws_size,
                              hipStream_t stream)
{
    (void)in_sizes; (void)n_in; (void)out_size; (void)ws_size;

    const float* xyz0  = (const float*)d_in[0];
    const float* feat0 = (const float*)d_in[1];
    const float* Wl[4]     = {(const float*)d_in[2], (const float*)d_in[5],
                              (const float*)d_in[8], (const float*)d_in[11]};
    const float* gammal[4] = {(const float*)d_in[3], (const float*)d_in[6],
                              (const float*)d_in[9], (const float*)d_in[12]};
    const float* betal[4]  = {(const float*)d_in[4], (const float*)d_in[7],
                              (const float*)d_in[10], (const float*)d_in[13]};

    static const int   Ns[5] = {65536, 65526, 32768, 16384, 8192};
    static const int   Cs[5] = {3, 64, 128, 256, 256};
    static const int   Sl[4] = {32, 32, 64, 64};
    static const float Rl[4] = {0.02f, 0.04f, 0.08f, 0.12f};

    size_t xyz_off[5], feat_off[5], o = 0;
    for (int l = 0; l < 5; ++l) { xyz_off[l]  = o; o += (size_t)Ns[l] * 3; }
    for (int l = 0; l < 5; ++l) { feat_off[l] = o; o += (size_t)Cs[l] * Ns[l]; }

    float* out = (float*)d_out;

    // Pass-through outputs: l_xyz[0] and l_feat[0].
    hipMemcpyAsync(out + xyz_off[0], xyz0, sizeof(float) * (size_t)Ns[0] * 3,
                   hipMemcpyDeviceToDevice, stream);
    hipMemcpyAsync(out + feat_off[0], feat0, sizeof(float) * (size_t)Cs[0] * Ns[0],
                   hipMemcpyDeviceToDevice, stream);

    // Workspace: dist (64K f32) | fidx (64K i32) | gidx (2.1M i32) | wpack
    float*    dist  = (float*)d_ws;
    int*      fidx  = (int*)(dist + 65536);
    int*      gidx  = fidx + 65536;
    unsigned* wpack = (unsigned*)(gidx + 2096832);   // 16B-aligned offset

    for (int l = 0; l < 4; ++l) {
        const int N    = Ns[l];
        const int P    = Ns[l + 1];
        const int S    = Sl[l];
        const int Cin  = Cs[l] + 3;
        const int Kpad = (Cin + 31) & ~31;
        const int nkt  = Kpad >> 5;
        const int O    = Cs[l + 1];

        const float* lx = out + xyz_off[l];
        const float* lf = out + feat_off[l];
        float*       nx = out + xyz_off[l + 1];
        float*       nf = out + feat_off[l + 1];

        fps_kernel<<<1, 1024, 0, stream>>>(lx, N, P, dist, fidx);
        gather_xyz_kernel<<<(P + 255) / 256, 256, 0, stream>>>(lx, fidx, nx, P);
        ball_query_kernel<<<(P + 127) / 128, 128, 0, stream>>>(
            nx, lx, P, N, Rl[l] * Rl[l], S, gidx);
        zero_kernel<<<512, 256, 0, stream>>>(nf, (size_t)O * P);

        const int wtotal = (O >> 4) * nkt * 256;
        pack_w_kernel<<<(wtotal + 255) / 256, 256, 0, stream>>>(
            Wl[l], wpack, wtotal, Cin, nkt);

        const int ntiles = (P * S) / 16;
        const int MT = O / 64;
        if (MT == 1)
            sa_gemm_pool_kernel<1><<<ntiles, 128, 0, stream>>>(
                wpack, gammal[l], betal[l], lx, nx, lf, gidx, nf, Cin, nkt, P, S, N);
        else if (MT == 2)
            sa_gemm_pool_kernel<2><<<ntiles, 128, 0, stream>>>(
                wpack, gammal[l], betal[l], lx, nx, lf, gidx, nf, Cin, nkt, P, S, N);
        else
            sa_gemm_pool_kernel<4><<<ntiles, 128, 0, stream>>>(
                wpack, gammal[l], betal[l], lx, nx, lf, gidx, nf, Cin, nkt, P, S, N);
    }
}
